// HierarchicalMemoryMoE_59098749993487
// MI455X (gfx1250) — matrix operand order repaired and checked by reading, not compiled, not run on hardware
//
#include <hip/hip_runtime.h>
#include <stdint.h>

// ---------------------------------------------------------------- constants
static constexpr int kE   = 4;
static constexpr int kN   = 2048;
static constexpr int kD   = 256;
static constexpr int kH   = 8;
static constexpr int kHD  = 32;       // head dim
static constexpr int kF   = 2048;
static constexpr int kM   = 4096;     // memory slots
static constexpr int kL   = 3;
static constexpr int kCTX = kE * kN;  // 8192
static constexpr int kPadQK = kH * 64; // 512: per-head K padded 32->64

typedef _Float16 h8   __attribute__((ext_vector_type(8)));
typedef _Float16 v16h __attribute__((ext_vector_type(16)));
typedef float    v8f  __attribute__((ext_vector_type(8)));

#define BM 128
#define BN 64
#define BK 64

// ---------------------------------------------------------------- WMMA GEMM (NT, async, double-buffered)
// C[M,N] = alpha * A[M,K] @ B[N,K]^T (+bias[n]) (+resid[m,n]) (relu?)
// Requirements (guaranteed by host): M%128==0, K%64==0; B readable for
// ceil(N/64)*64 rows (pad rows may be garbage -> lands only in unstored cols).
// Tiles staged with global_load_async_to_lds_b128 (ASYNCcnt), ping-pong LDS.
__global__ void __launch_bounds__(256)
gemm_wmma_nt(int Mdim, int Ndim, int Kdim,
             const _Float16* __restrict__ A, int lda,
             const _Float16* __restrict__ B, int ldb,
             float* __restrict__ Cf, _Float16* __restrict__ Ch, int ldc,
             float alpha,
             const float* __restrict__ bias,
             const float* __restrict__ resid,
             int doRelu)
{
    __shared__ _Float16 smem[2 * (BM * BK + BN * BK)];   // 48 KB
    _Float16* Asb = smem;                     // [2][BM][BK]
    _Float16* Bsb = smem + 2 * BM * BK;       // [2][BN][BK] (k contiguous)

    const int tid  = threadIdx.x;
    const int lane = tid & 31;
    const int wid  = tid >> 5;
    const int wm   = (wid >> 1) * 32;   // 0,32,64,96
    const int wn   = (wid & 1) * 32;    // 0,32
    const int r    = lane & 15;
    const int hi   = lane >> 4;
    const int m0   = blockIdx.y * BM;
    const int n0   = blockIdx.x * BN;

    v8f acc[2][2] = {};

    const unsigned ldsA = (unsigned)(uintptr_t)Asb;   // LDS aperture: low 32b = LDS byte addr
    const unsigned ldsB = (unsigned)(uintptr_t)Bsb;

    auto issue = [&](int k0, int buf) {
        #pragma unroll
        for (int t = 0; t < 4; ++t) {                 // A tile: 1024 h8 vectors
            int v = tid + t * 256;
            int row = v >> 3, kc = (v & 7) * 8;
            unsigned l = ldsA + (unsigned)(((buf * BM + row) * BK + kc) * 2);
            uint64_t g = (uint64_t)(uintptr_t)(A + (size_t)(m0 + row) * lda + k0 + kc);
            asm volatile("global_load_async_to_lds_b128 %0, %1, off"
                         :: "v"(l), "v"(g) : "memory");
        }
        #pragma unroll
        for (int t = 0; t < 2; ++t) {                 // B tile: 512 h8 vectors
            int v = tid + t * 256;
            int n = v >> 3, kc = (v & 7) * 8;
            unsigned l = ldsB + (unsigned)(((buf * BN + n) * BK + kc) * 2);
            uint64_t g = (uint64_t)(uintptr_t)(B + (size_t)(n0 + n) * ldb + k0 + kc);
            asm volatile("global_load_async_to_lds_b128 %0, %1, off"
                         :: "v"(l), "v"(g) : "memory");
        }
    };

    auto compute = [&](int buf) {
        const _Float16* As = Asb + buf * BM * BK;
        const _Float16* Bs = Bsb + buf * BN * BK;
        #pragma unroll
        for (int kk = 0; kk < BK; kk += 32) {
            v16h a[2], b[2];
            // A frag (ISA 16-bit A 16x32): elems0-7 K=hi*8+q, elems8-15 K=16+hi*8+q
            #pragma unroll
            for (int i = 0; i < 2; ++i) {
                h8 lo = *(const h8*)(As + (wm + i * 16 + r) * BK + kk + hi * 8);
                h8 hh = *(const h8*)(As + (wm + i * 16 + r) * BK + kk + 16 + hi * 8);
                #pragma unroll
                for (int q = 0; q < 8; ++q) { a[i][q] = lo[q]; a[i][q + 8] = hh[q]; }
            }
            // B frag: col n=r, elems0-7 K=hi*8+q, elems8-15 K=16+hi*8+q
            #pragma unroll
            for (int j = 0; j < 2; ++j) {
                h8 lo = *(const h8*)(Bs + (wn + j * 16 + r) * BK + kk + hi * 8);
                h8 hh = *(const h8*)(Bs + (wn + j * 16 + r) * BK + kk + 16 + hi * 8);
                #pragma unroll
                for (int q = 0; q < 8; ++q) { b[j][q] = lo[q]; b[j][q + 8] = hh[q]; }
            }
            #pragma unroll
            for (int i = 0; i < 2; ++i)
                #pragma unroll
                for (int j = 0; j < 2; ++j)
                    acc[i][j] = __builtin_amdgcn_wmma_f32_16x16x32_f16(
                        false, a[i], false, b[j], (short)0, acc[i][j], false, false);
        }
    };

    const int T = Kdim / BK;
    issue(0, 0);
    for (int t = 0; t < T; ++t) {
        int cur = t & 1;
        asm volatile("s_wait_asynccnt 0x0" ::: "memory");  // tile t landed (this wave)
        __syncthreads();                                   // visible to all; prev reads done
        if (t + 1 < T) issue((t + 1) * BK, cur ^ 1);       // prefetch next tile during compute
        compute(cur);
        __syncthreads();                                   // all done reading buf before overwrite
    }

    // ---- epilogue: C layout lane r = col, vgpr v -> row v + 8*hi
    #pragma unroll
    for (int i = 0; i < 2; ++i) {
        #pragma unroll
        for (int j = 0; j < 2; ++j) {
            #pragma unroll
            for (int v = 0; v < 8; ++v) {
                int gm = m0 + wm + i * 16 + hi * 8 + v;
                int gn = n0 + wn + j * 16 + r;
                if (gm < Mdim && gn < Ndim) {
                    float val = acc[i][j][v] * alpha;
                    if (bias)  val += bias[gn];
                    if (resid) val += resid[(size_t)gm * ldc + gn];
                    if (doRelu) val = fmaxf(val, 0.f);
                    if (Cf) Cf[(size_t)gm * ldc + gn] = val;
                    if (Ch) Ch[(size_t)gm * ldc + gn] = (_Float16)val;
                }
            }
        }
    }
}

// ---------------------------------------------------------------- transpose
// dst[c][r] = src[r][c]; R,C multiples of 64. Tiled 64x64 via padded LDS.
__global__ void __launch_bounds__(256)
transpose16(const _Float16* __restrict__ src, int srcLd,
            _Float16* __restrict__ dst, int dstLd)
{
    __shared__ _Float16 t[64][65];
    const int c0 = blockIdx.x * 64, r0 = blockIdx.y * 64;
    const int tid = threadIdx.x;
    #pragma unroll
    for (int it = 0; it < 16; ++it) {
        int idx = it * 256 + tid;
        int row = idx >> 6, col = idx & 63;
        t[row][col] = src[(size_t)(r0 + row) * srcLd + c0 + col];
    }
    __syncthreads();
    #pragma unroll
    for (int it = 0; it < 16; ++it) {
        int idx = it * 256 + tid;
        int row = idx >> 6, col = idx & 63;
        dst[(size_t)(c0 + row) * dstLd + r0 + col] = t[col][row];
    }
}

// build per-head K=64 zero-padded q/k: qpad[n][h*64+j], kpad[n][h*64+j]
__global__ void padqk_kernel(const _Float16* __restrict__ qkv,
                             _Float16* __restrict__ qpad, _Float16* __restrict__ kpad)
{
    int i = blockIdx.x * 256 + threadIdx.x;    // over kN*kPadQK
    int n = i >> 9, c = i & (kPadQK - 1);
    int h = c >> 6, j = c & 63;
    _Float16 qv = (j < kHD) ? qkv[(size_t)n * 3 * kD + h * kHD + j] : (_Float16)0;
    _Float16 kv = (j < kHD) ? qkv[(size_t)n * 3 * kD + kD + h * kHD + j] : (_Float16)0;
    qpad[(size_t)n * kPadQK + c] = qv;
    kpad[(size_t)n * kPadQK + c] = kv;
}

// ---------------------------------------------------------------- helpers
__global__ void cvt_f32_f16(const float* __restrict__ x, _Float16* __restrict__ y, size_t n)
{
    size_t i = (size_t)blockIdx.x * 256 + threadIdx.x;
    if (i < n) y[i] = (_Float16)x[i];
}

__global__ void zero_f32(float* __restrict__ x, size_t n)
{
    size_t i = (size_t)blockIdx.x * 256 + threadIdx.x;
    if (i < n) x[i] = 0.f;
}

// row softmax, f32 in -> f16 probs out; columns in [maskLo,maskHi) forced to 0
__global__ void __launch_bounds__(256)
softmax_rows(const float* __restrict__ S, _Float16* __restrict__ P,
             int cols, int maskLo, int maskHi)
{
    __shared__ float red[256];
    const int row = blockIdx.x;
    const int tid = threadIdx.x;
    const float* s = S + (size_t)row * cols;
    _Float16*    p = P + (size_t)row * cols;

    float mx = -3.4e38f;
    for (int c = tid; c < cols; c += 256)
        if (c < maskLo || c >= maskHi) mx = fmaxf(mx, s[c]);
    red[tid] = mx; __syncthreads();
    for (int o = 128; o > 0; o >>= 1) {
        if (tid < o) red[tid] = fmaxf(red[tid], red[tid + o]);
        __syncthreads();
    }
    mx = red[0]; __syncthreads();

    float sum = 0.f;
    for (int c = tid; c < cols; c += 256) {
        bool valid = (c < maskLo) || (c >= maskHi);
        float e = valid ? __expf(s[c] - mx) : 0.f;
        p[c] = (_Float16)e;
        sum += e;
    }
    red[tid] = sum; __syncthreads();
    for (int o = 128; o > 0; o >>= 1) {
        if (tid < o) red[tid] += red[tid + o];
        __syncthreads();
    }
    float inv = 1.f / red[0]; __syncthreads();

    for (int c = tid; c < cols; c += 256)
        p[c] = (_Float16)((float)p[c] * inv);
}

// y = LN(z) * gamma + beta ; row length 256, dual f32/f16 output
__global__ void __launch_bounds__(256)
ln_rows(const float* __restrict__ z, const float* __restrict__ gamma,
        const float* __restrict__ beta, float* __restrict__ yf,
        _Float16* __restrict__ yh)
{
    __shared__ float red[256];
    const int row = blockIdx.x;
    const int d = threadIdx.x;
    float v = z[(size_t)row * kD + d];
    red[d] = v; __syncthreads();
    for (int o = 128; o > 0; o >>= 1) { if (d < o) red[d] += red[d + o]; __syncthreads(); }
    float mu = red[0] * (1.f / kD); __syncthreads();
    float dv = v - mu;
    red[d] = dv * dv; __syncthreads();
    for (int o = 128; o > 0; o >>= 1) { if (d < o) red[d] += red[d + o]; __syncthreads(); }
    float var = red[0] * (1.f / kD);
    float y = dv * rsqrtf(var + 1e-5f) * gamma[d] + beta[d];
    yf[(size_t)row * kD + d] = y;
    yh[(size_t)row * kD + d] = (_Float16)y;
}

__global__ void ema_kernel(const float* __restrict__ mem, const float* __restrict__ upds,
                           float* __restrict__ out, size_t n)
{
    size_t i = (size_t)blockIdx.x * 256 + threadIdx.x;
    if (i < n) out[i] = 0.9f * mem[i] + 0.1f * upds[i];
}

__global__ void __launch_bounds__(256)
pool_kernel(const float* __restrict__ eout, float* __restrict__ pooled)
{
    const int e = blockIdx.x, d = threadIdx.x;
    const float* p = eout + (size_t)e * kN * kD + d;
    float s = 0.f;
    for (int n = 0; n < kN; ++n) s += p[(size_t)n * kD];
    pooled[e * kD + d] = s * (1.f / kN);
}

__global__ void __launch_bounds__(256)
gate_kernel(const float* __restrict__ pooled, const float* __restrict__ gg,
            float* __restrict__ gatesWS, float* __restrict__ gatesOut)
{
    __shared__ float red[256];
    __shared__ float logits[kE];
    const int d = threadIdx.x;
    for (int e = 0; e < kE; ++e) {
        red[d] = pooled[e * kD + d] * gg[d];
        __syncthreads();
        for (int o = 128; o > 0; o >>= 1) { if (d < o) red[d] += red[d + o]; __syncthreads(); }
        if (d == 0) logits[e] = red[0];
        __syncthreads();
    }
    if (d == 0) {
        float mx = fmaxf(fmaxf(logits[0], logits[1]), fmaxf(logits[2], logits[3]));
        float ex[kE], s = 0.f;
        for (int e = 0; e < kE; ++e) { ex[e] = __expf(logits[e] - mx); s += ex[e]; }
        for (int e = 0; e < kE; ++e) {
            float g = ex[e] / s;
            gatesWS[e] = g;
            gatesOut[e] = g;
        }
    }
}

__global__ void fuse_kernel(const float* __restrict__ eout, const float* __restrict__ gates,
                            float* __restrict__ out, int nd)
{
    int i = blockIdx.x * 256 + threadIdx.x;
    if (i < nd) {
        float s = 0.f;
        #pragma unroll
        for (int e = 0; e < kE; ++e) s += gates[e] * eout[(size_t)e * nd + i];
        out[i] = s;
    }
}

// ---------------------------------------------------------------- host side
static inline unsigned cdiv(size_t a, size_t b) { return (unsigned)((a + b - 1) / b); }

static void run_gemm(int M_, int N_, int K_,
                     const _Float16* A, int lda, const _Float16* B, int ldb,
                     float* Cf, _Float16* Ch, int ldc, float alpha,
                     const float* bias, const float* resid, int relu, hipStream_t s)
{
    dim3 g(cdiv(N_, BN), cdiv(M_, BM)), b(256);
    gemm_wmma_nt<<<g, b, 0, s>>>(M_, N_, K_, A, lda, B, ldb, Cf, Ch, ldc,
                                 alpha, bias, resid, relu);
}

static void run_transpose(const _Float16* src, int srcLd, _Float16* dst, int dstLd,
                          int R, int C, hipStream_t s)
{
    dim3 g(C / 64, R / 64), b(256);
    transpose16<<<g, b, 0, s>>>(src, srcLd, dst, dstLd);
}

extern "C" void kernel_launch(void* const* d_in, const int* in_sizes, int n_in,
                              void* d_out, int out_size, void* d_ws, size_t ws_size,
                              hipStream_t stream)
{
    (void)in_sizes; (void)n_in; (void)out_size; (void)ws_size;
    const float* tokens    = (const float*)d_in[0];
    const float* memories  = (const float*)d_in[1];
    const float* in_proj_w = (const float*)d_in[2];
    const float* in_proj_b = (const float*)d_in[3];
    const float* out_w     = (const float*)d_in[4];
    const float* out_b     = (const float*)d_in[5];
    const float* lin1_w    = (const float*)d_in[6];
    const float* lin1_b    = (const float*)d_in[7];
    const float* lin2_w    = (const float*)d_in[8];
    const float* lin2_b    = (const float*)d_in[9];
    const float* ln1_w     = (const float*)d_in[10];
    const float* ln1_b     = (const float*)d_in[11];
    const float* ln2_w     = (const float*)d_in[12];
    const float* ln2_b     = (const float*)d_in[13];
    const float* agg_w     = (const float*)d_in[14];
    const float* agg_b     = (const float*)d_in[15];
    const float* gg        = (const float*)d_in[16];

    float* out_fused = (float*)d_out;                    // [N*D]
    float* out_gates = out_fused + (size_t)kN * kD;      // [E]
    float* out_mem   = out_gates + kE;                   // [L*M*D]

    // ---- workspace carve
    char* wsp = (char*)d_ws;
    size_t off = 0;
    auto A16 = [&](size_t n) { _Float16* p = (_Float16*)(wsp + off); off += ((n * 2 + 255) / 256) * 256; return p; };
    auto A32 = [&](size_t n) { float*    p = (float*)   (wsp + off); off += ((n * 4 + 255) / 256) * 256; return p; };

    _Float16* tok16  = A16((size_t)kE * kN * kD);
    _Float16* tokT16 = A16((size_t)kD * kCTX);            // tokens^T [D][CTX]
    _Float16* mem16  = A16((size_t)kL * kM * kD);
    _Float16* memT16 = A16((size_t)kL * kD * kM);         // per-layer mem^T [D][M]
    _Float16* wqkv16 = A16((size_t)kE * 3 * kD * kD);
    _Float16* wo16   = A16((size_t)kE * kD * kD);
    _Float16* w116   = A16((size_t)kE * kF * kD);
    _Float16* w216   = A16((size_t)kE * kD * kF);
    _Float16* agg16  = A16((size_t)kD * kL * kD);
    _Float16* P16    = A16((size_t)kN * kCTX);
    _Float16* PT16   = A16((size_t)kM * kN);              // P^T for update GEMM
    _Float16* x16    = A16((size_t)kN * kD);
    _Float16* xT16   = A16((size_t)kD * kN);              // x^T [D][N]
    _Float16* qkv16  = A16((size_t)kN * 3 * kD);
    _Float16* qpad   = A16((size_t)kN * kPadQK);          // q, K padded 32->64 per head
    _Float16* kpad   = A16((size_t)kN * kPadQK);
    _Float16* vT16   = A16((size_t)(kD + BN) * kN);       // v^T [D][N] + 64 pad rows
    _Float16* att16  = A16((size_t)kN * kD);
    _Float16* h16    = A16((size_t)kN * kF);
    _Float16* ret16  = A16((size_t)kN * kL * kD);
    float* S    = A32((size_t)kN * kCTX);
    float* xf   = A32((size_t)kN * kD);
    float* zf   = A32((size_t)kN * kD);
    float* upds = A32((size_t)kL * kM * kD);
    float* eout = A32((size_t)kE * kN * kD);
    float* pooled  = A32((size_t)kE * kD);
    float* gatesWS = A32(kE);

    auto cvt = [&](const float* src, _Float16* dst, size_t n) {
        cvt_f32_f16<<<cdiv(n, 256), 256, 0, stream>>>(src, dst, n);
    };
    cvt(tokens,    tok16,  (size_t)kE * kN * kD);
    cvt(memories,  mem16,  (size_t)kL * kM * kD);
    cvt(in_proj_w, wqkv16, (size_t)kE * 3 * kD * kD);
    cvt(out_w,     wo16,   (size_t)kE * kD * kD);
    cvt(lin1_w,    w116,   (size_t)kE * kF * kD);
    cvt(lin2_w,    w216,   (size_t)kE * kD * kF);
    cvt(agg_w,     agg16,  (size_t)kD * kL * kD);

    // global transposes (once per call)
    run_transpose(tok16, kD, tokT16, kCTX, kCTX, kD, stream);         // [CTX,D] -> [D,CTX]
    for (int l = 0; l < kL; ++l)
        run_transpose(mem16 + (size_t)l * kM * kD, kD,
                      memT16 + (size_t)l * kD * kM, kM, kM, kD, stream);

    zero_f32<<<cdiv((size_t)kL * kM * kD, 256), 256, 0, stream>>>(upds, (size_t)kL * kM * kD);

    const float scaleD  = 0.0625f;               // 1/sqrt(256)
    const float scaleHD = 0.1767766953f;         // 1/sqrt(32)

    for (int e = 0; e < kE; ++e) {
        const _Float16* tokE16 = tok16 + (size_t)e * kN * kD;
        const float*    tokE   = tokens + (size_t)e * kN * kD;

        // ---- cross-expert attention (self block masked in softmax)
        run_gemm(kN, kCTX, kD, tokE16, kD, tok16, kD,
                 S, nullptr, kCTX, scaleD, nullptr, nullptr, 0, stream);
        softmax_rows<<<kN, 256, 0, stream>>>(S, P16, kCTX, e * kN, (e + 1) * kN);
        run_gemm(kN, kD, kCTX, P16, kCTX, tokT16, kCTX,
                 xf, x16, kD, 1.f, nullptr, tokE, 0, stream);   // x = tok + P @ tokens_all

        // ---- MHA: qkv projection, per-head padded layouts
        run_gemm(kN, 3 * kD, kD, x16, kD, wqkv16 + (size_t)e * 3 * kD * kD, kD,
                 nullptr, qkv16, 3 * kD, 1.f, in_proj_b + (size_t)e * 3 * kD, nullptr, 0, stream);
        padqk_kernel<<<cdiv((size_t)kN * kPadQK, 256), 256, 0, stream>>>(qkv16, qpad, kpad);
        run_transpose(qkv16 + 2 * kD, 3 * kD, vT16, kN, kN, kD, stream);  // v^T [D][N]
        for (int h = 0; h < kH; ++h) {
            run_gemm(kN, kN, 64, qpad + h * 64, kPadQK, kpad + h * 64, kPadQK,
                     S, nullptr, kN, scaleHD, nullptr, nullptr, 0, stream);
            softmax_rows<<<kN, 256, 0, stream>>>(S, P16, kN, 0, 0);
            // o_h = P @ v_h = P[N,K] @ (vT_h[N=32(+pad),K])^T ; pad cols unstored
            run_gemm(kN, kHD, kN, P16, kN, vT16 + (size_t)h * kHD * kN, kN,
                     nullptr, att16 + h * kHD, kD, 1.f, nullptr, nullptr, 0, stream);
        }
        // out proj + residual -> z ; LN1 -> x
        run_gemm(kN, kD, kD, att16, kD, wo16 + (size_t)e * kD * kD, kD,
                 zf, nullptr, kD, 1.f, out_b + (size_t)e * kD, xf, 0, stream);
        ln_rows<<<kN, 256, 0, stream>>>(zf, ln1_w + (size_t)e * kD, ln1_b + (size_t)e * kD, xf, x16);

        // ---- FFN
        run_gemm(kN, kF, kD, x16, kD, w116 + (size_t)e * kF * kD, kD,
                 nullptr, h16, kF, 1.f, lin1_b + (size_t)e * kF, nullptr, 1, stream);
        run_gemm(kN, kD, kF, h16, kF, w216 + (size_t)e * kD * kF, kF,
                 zf, nullptr, kD, 1.f, lin2_b + (size_t)e * kD, xf, 0, stream);
        ln_rows<<<kN, 256, 0, stream>>>(zf, ln2_w + (size_t)e * kD, ln2_b + (size_t)e * kD, xf, x16);
        run_transpose(x16, kD, xT16, kN, kN, kD, stream);                 // x^T for update GEMM

        // ---- memory read + update accumulation
        for (int l = 0; l < kL; ++l) {
            const _Float16* memL  = mem16  + (size_t)l * kM * kD;
            const _Float16* memTL = memT16 + (size_t)l * kD * kM;
            run_gemm(kN, kM, kD, x16, kD, memL, kD,
                     S, nullptr, kM, scaleD, nullptr, nullptr, 0, stream);
            softmax_rows<<<kN, 256, 0, stream>>>(S, P16, kM, 0, 0);
            run_gemm(kN, kD, kM, P16, kM, memTL, kM,
                     nullptr, ret16 + l * kD, kL * kD, 1.f, nullptr, nullptr, 0, stream);
            run_transpose(P16, kM, PT16, kN, kN, kM, stream);             // P^T [M][N]
            float* updL = upds + (size_t)l * kM * kD;
            run_gemm(kM, kD, kN, PT16, kN, xT16, kN,
                     updL, nullptr, kD, 1.f, nullptr, updL, 0, stream);   // += P^T @ x
        }
        // ---- aggregate memory reads + residual -> expert output
        run_gemm(kN, kD, kL * kD, ret16, kL * kD, agg16, kL * kD,
                 eout + (size_t)e * kN * kD, nullptr, kD, 1.f, agg_b, xf, 0, stream);
    }

    // ---- EMA memory update, gated fusion
    ema_kernel<<<cdiv((size_t)kL * kM * kD, 256), 256, 0, stream>>>(
        memories, upds, out_mem, (size_t)kL * kM * kD);
    pool_kernel<<<kE, 256, 0, stream>>>(eout, pooled);
    gate_kernel<<<1, 256, 0, stream>>>(pooled, gg, gatesWS, out_gates);
    fuse_kernel<<<cdiv((size_t)kN * kD, 256), 256, 0, stream>>>(
        eout, gatesWS, out_fused, kN * kD);
}
